// MultiModalGAT_69784628625409
// MI455X (gfx1250) — matrix operand order repaired and checked by reading, not compiled, not run on hardware
//
#include <hip/hip_runtime.h>
#include <hip/hip_bf16.h>
#include <math.h>

// ---------------------------------------------------------------------------
// MultiModal GATv2 for MI455X (gfx1250).
// GEMMs (≈380 GFLOP) run on v_wmma_f32_16x16x32_bf16, wave32. Each wave owns a
// 32x32 output tile (2 A frags x 2 B frags -> 4 WMMAs per 8 b128 loads), which
// doubles L2-byte reuse vs a 16x16 tile. Weights are transposed to bf16 via an
// LDS-tiled transpose so both WMMA operands load as contiguous b128 pairs.
// Reuse is served by the 192MB L2 (weights 8MB bf16, activations 40MB bf16).
// Edge softmax/aggregation phase is memory-bound f32 with atomics.
// ---------------------------------------------------------------------------

#define NODES 10000
#define NPAD  10016            // NODES rounded up to 32 for the GEMM M-tiling
#define EDGES 40000
#define NEDGE (EDGES + NODES)  // self loops appended
#define HC    2048             // HEADS * HID
#define NHEAD 8
#define HID   256

typedef __attribute__((ext_vector_type(16))) __bf16       v16bf;
typedef __attribute__((ext_vector_type(8)))  float        v8f;
typedef __attribute__((ext_vector_type(4)))  unsigned int u32x4;

// ------------------------- small utility kernels ---------------------------

__global__ void __launch_bounds__(256)
k_build_edges(const int* __restrict__ ei, int* __restrict__ srcA, int* __restrict__ dstA) {
    int i = blockIdx.x * 256 + threadIdx.x;
    if (i >= NEDGE) return;
    if (i < EDGES) { srcA[i] = ei[i]; dstA[i] = ei[EDGES + i]; }
    else           { srcA[i] = i - EDGES; dstA[i] = i - EDGES; }
}

__global__ void __launch_bounds__(256)
k_cvt_bf16(const float* __restrict__ in, __bf16* __restrict__ out, int n) {
    int i = blockIdx.x * 256 + threadIdx.x;
    if (i < n) out[i] = (__bf16)in[i];
}

// W f32 [K][2048] -> BT bf16 [2048][K], LDS-tiled 32x32 transpose:
// coalesced f32 reads along HC, coalesced bf16 writes along K.
__global__ void __launch_bounds__(256)
k_cvtT_bf16(const float* __restrict__ W, __bf16* __restrict__ BT, int K) {
    __shared__ __bf16 tile[32][33];                 // +1 pad kills bank conflicts
    const int c0 = blockIdx.x * 32;                 // col block in W
    const int k0 = blockIdx.y * 32;                 // row block in W
    const int tx = threadIdx.x;                     // 0..31
    const int ty = threadIdx.y;                     // 0..7
#pragma unroll
    for (int i = ty; i < 32; i += 8)
        tile[i][tx] = (__bf16)W[(size_t)(k0 + i) * HC + c0 + tx];
    __syncthreads();
#pragma unroll
    for (int i = ty; i < 32; i += 8)                // BT[c0+i][k0+tx] = W[k0+tx][c0+i]
        BT[(size_t)(c0 + i) * K + k0 + tx] = tile[tx][i];
}

__global__ void __launch_bounds__(256)
k_fill_f32(float* __restrict__ p, float v, int n) {
    int i = blockIdx.x * 256 + threadIdx.x;
    if (i < n) p[i] = v;
}

// ------------------------------ WMMA GEMM ----------------------------------
// C[M][Nc] = A[Mpad][K](bf16,row-major) * BT[Nc][K](bf16, pre-transposed) + bias
// 32x32 tile per wave: frags a0/a1 (rows m0..+15 / +16..+31), b0/b1 (cols),
// 4 accumulators, 4 v_wmma per 8 b128 loads (2 loads/wmma, 16 FLOP/L2-byte).
// A fragment per lane (row = base+lane&15, half = lane>>4):
//   elems 0..7  = K[32t + 8*half ..], elems 8..15 = K[32t + 16 + 8*half ..]
// B fragment per lane (col = base+lane&15): elems 0..7  = K[32t + 8*half ..], elems 8..15 = K[32t + 16 + 8*half ..]

__global__ void __launch_bounds__(256)
k_gemm_bf16(const __bf16* __restrict__ A, const __bf16* __restrict__ BT,
            const float* __restrict__ bias, float* __restrict__ C,
            int M, int Mtiles, int K, int Nc) {
    const int lane = threadIdx.x & 31;
    const int wave = threadIdx.x >> 5;
    const int ntn  = Nc >> 5;                       // 32-wide col tiles
    const int tile = blockIdx.x * 8 + wave;
    if (tile >= Mtiles * ntn) return;
    const int m0   = (tile / ntn) << 5;
    const int n0   = (tile % ntn) << 5;
    const int half = lane >> 4;                     // 0 or 1
    const int l16  = lane & 15;

    const __bf16* a0 = A  + (size_t)(m0 + l16) * K + half * 8;
    const __bf16* a1 = a0 + (size_t)16 * K;
    const __bf16* b0 = BT + (size_t)(n0 + l16) * K + half * 8;
    const __bf16* b1 = b0 + (size_t)16 * K;

    v8f acc00 = {}, acc01 = {}, acc10 = {}, acc11 = {};
    union F { v16bf v; u32x4 u[2]; };
    for (int k = 0; k < K; k += 32) {
        F fa0, fa1, fb0, fb1;
        fa0.u[0] = *reinterpret_cast<const u32x4*>(a0 + k);
        fa0.u[1] = *reinterpret_cast<const u32x4*>(a0 + k + 16);
        fa1.u[0] = *reinterpret_cast<const u32x4*>(a1 + k);
        fa1.u[1] = *reinterpret_cast<const u32x4*>(a1 + k + 16);
        fb0.u[0] = *reinterpret_cast<const u32x4*>(b0 + k);
        fb0.u[1] = *reinterpret_cast<const u32x4*>(b0 + k + 16);
        fb1.u[0] = *reinterpret_cast<const u32x4*>(b1 + k);
        fb1.u[1] = *reinterpret_cast<const u32x4*>(b1 + k + 16);
        acc00 = __builtin_amdgcn_wmma_f32_16x16x32_bf16(false, fa0.v, false, fb0.v, (short)0, acc00, false, false);
        acc01 = __builtin_amdgcn_wmma_f32_16x16x32_bf16(false, fa0.v, false, fb1.v, (short)0, acc01, false, false);
        acc10 = __builtin_amdgcn_wmma_f32_16x16x32_bf16(false, fa1.v, false, fb0.v, (short)0, acc10, false, false);
        acc11 = __builtin_amdgcn_wmma_f32_16x16x32_bf16(false, fa1.v, false, fb1.v, (short)0, acc11, false, false);
    }

    const int   col0 = n0 + l16, col1 = col0 + 16;
    const float bv0  = bias[col0], bv1 = bias[col1];
    float* c0 = C + (size_t)(m0 + half * 8) * Nc;   // rows m0+8*half+v
#pragma unroll
    for (int v = 0; v < 8; ++v) {
        c0[(size_t)v * Nc + col0] = acc00[v] + bv0;
        c0[(size_t)v * Nc + col1] = acc01[v] + bv1;
    }
    if (m0 + 16 < M) {                              // upper 16-row subtile (pad guard)
        float* c1 = c0 + (size_t)16 * Nc;
#pragma unroll
        for (int v = 0; v < 8; ++v) {
            c1[(size_t)v * Nc + col0] = acc10[v] + bv0;
            c1[(size_t)v * Nc + col1] = acc11[v] + bv1;
        }
    }
}

// ---------------------------- edge-phase kernels ---------------------------

__device__ __forceinline__ void atomicMaxF(float* addr, float val) {
    if (val >= 0.0f) atomicMax((int*)addr, __float_as_int(val));
    else             atomicMin((unsigned int*)addr, __float_as_uint(val));
}

// one wave per (edge, head): score = att_h . LeakyReLU(hl[src,h,:] + hr[dst,h,:])
__global__ void __launch_bounds__(256)
k_edge_score(const float* __restrict__ hl, const float* __restrict__ hr,
             const float* __restrict__ att, const int* __restrict__ srcA,
             const int* __restrict__ dstA, float* __restrict__ score,
             float* __restrict__ smax) {
    int gid  = blockIdx.x * 256 + threadIdx.x;
    int wid  = gid >> 5;
    int lane = gid & 31;
    if (wid >= NEDGE * NHEAD) return;
    int e = wid >> 3, h = wid & 7;
    int s = srcA[e], d = dstA[e];
    const float* pl = hl + (size_t)s * HC + h * HID;
    const float* pr = hr + (size_t)d * HC + h * HID;
    const float* pa = att + h * HID;
    float acc = 0.0f;
    for (int c = lane; c < HID; c += 32) {
        float g = pl[c] + pr[c];
        g = (g > 0.0f) ? g : 0.2f * g;        // LeakyReLU(0.2)
        acc += g * pa[c];
    }
#pragma unroll
    for (int off = 16; off > 0; off >>= 1)
        acc += __shfl_xor(acc, off, 32);
    if (lane == 0) {
        score[wid] = acc;
        atomicMaxF(&smax[d * NHEAD + h], acc);
    }
}

__global__ void __launch_bounds__(256)
k_edge_exp(const int* __restrict__ dstA, float* __restrict__ score,
           const float* __restrict__ smax, float* __restrict__ ssum) {
    int i = blockIdx.x * 256 + threadIdx.x;
    if (i >= NEDGE * NHEAD) return;
    int e = i >> 3, h = i & 7;
    int d = dstA[e];
    float a = __expf(score[i] - smax[d * NHEAD + h]);
    score[i] = a;                              // reuse as un-normalized alpha
    atomicAdd(&ssum[d * NHEAD + h], a);
}

// out[dst, j] += alpha(e, h=j>>8) * hl[src, j]
__global__ void __launch_bounds__(256)
k_aggregate(const float* __restrict__ hl, const int* __restrict__ srcA,
            const int* __restrict__ dstA, const float* __restrict__ score,
            const float* __restrict__ ssum, float* __restrict__ out) {
    int gid = blockIdx.x * 256 + threadIdx.x;
    if (gid >= NEDGE * HC) return;
    int e = gid >> 11;
    int j = gid & (HC - 1);
    int h = j >> 8;
    int s = srcA[e], d = dstA[e];
    float alpha = score[e * NHEAD + h] / (ssum[d * NHEAD + h] + 1e-16f);
    atomicAdd(&out[(size_t)d * HC + j], alpha * hl[(size_t)s * HC + j]);
}

__global__ void __launch_bounds__(256)
k_elu_bias_bf16(const float* __restrict__ agg, const float* __restrict__ bias,
                __bf16* __restrict__ XB) {
    int i = blockIdx.x * 256 + threadIdx.x;
    if (i >= NODES * HC) return;
    float v = agg[i] + bias[i & (HC - 1)];
    v = (v > 0.0f) ? v : expm1f(v);            // ELU(alpha=1)
    XB[i] = (__bf16)v;
}

__global__ void __launch_bounds__(256)
k_mean_heads(const float* __restrict__ agg, const float* __restrict__ b2,
             float* __restrict__ hfin) {
    int i = blockIdx.x * 256 + threadIdx.x;
    if (i >= NODES * HID) return;
    int n = i >> 8, c = i & (HID - 1);
    float s = 0.0f;
#pragma unroll
    for (int h = 0; h < NHEAD; ++h)
        s += agg[(size_t)n * HC + h * HID + c];
    hfin[i] = s * (1.0f / NHEAD) + b2[c];
}

// ------------------------------- head MLPs ---------------------------------

__global__ void __launch_bounds__(256)
k_linear(const float* __restrict__ in, const float* __restrict__ W,
         const float* __restrict__ b, float* __restrict__ out,
         int K, int J, int total, int do_relu) {
    int gid = blockIdx.x * 256 + threadIdx.x;
    if (gid >= total) return;
    int n = gid / J, j = gid - n * J;
    const float* ip = in + (size_t)n * K;
    float s = b[j];
    for (int k = 0; k < K; ++k) s += ip[k] * W[k * J + j];
    out[gid] = do_relu ? fmaxf(s, 0.0f) : s;
}

// ------------------------------- launcher ----------------------------------

extern "C" void kernel_launch(void* const* d_in, const int* in_sizes, int n_in,
                              void* d_out, int out_size, void* d_ws, size_t ws_size,
                              hipStream_t stream) {
    (void)in_sizes; (void)n_in; (void)out_size; (void)ws_size;
    const float* x  = (const float*)d_in[0];
    const int*   ei = (const int*)d_in[1];
    const float* Wl[3]  = {(const float*)d_in[2],  (const float*)d_in[8],  (const float*)d_in[14]};
    const float* bl[3]  = {(const float*)d_in[3],  (const float*)d_in[9],  (const float*)d_in[15]};
    const float* Wr[3]  = {(const float*)d_in[4],  (const float*)d_in[10], (const float*)d_in[16]};
    const float* br[3]  = {(const float*)d_in[5],  (const float*)d_in[11], (const float*)d_in[17]};
    const float* att[3] = {(const float*)d_in[6],  (const float*)d_in[12], (const float*)d_in[18]};
    const float* bb[3]  = {(const float*)d_in[7],  (const float*)d_in[13], (const float*)d_in[19]};
    const float* Wp1 = (const float*)d_in[20]; const float* bp1 = (const float*)d_in[21];
    const float* Wp2 = (const float*)d_in[22]; const float* bp2 = (const float*)d_in[23];
    const float* Ws1 = (const float*)d_in[24]; const float* bs1 = (const float*)d_in[25];
    const float* Ws2 = (const float*)d_in[26]; const float* bs2 = (const float*)d_in[27];
    float* outp = (float*)d_out;

    // ---- workspace carving (~320 MB) ----
    char* wsb = (char*)d_ws; size_t off = 0;
    auto alloc = [&](size_t bytes) -> void* {
        void* p = wsb + off; off += (bytes + 255) & ~(size_t)255; return p;
    };
    __bf16* XB   = (__bf16*)alloc((size_t)NPAD * HC * 2);    // bf16 activations (row-padded)
    __bf16* WT   = (__bf16*)alloc((size_t)HC * HC * 2);      // bf16 W^T (left)
    __bf16* WT2  = (__bf16*)alloc((size_t)HC * HC * 2);      // bf16 W^T (right)
    float*  hl   = (float*)alloc((size_t)NODES * HC * 4);
    float*  hr   = (float*)alloc((size_t)NODES * HC * 4);
    float*  agg  = (float*)alloc((size_t)NODES * HC * 4);
    int*    srcA = (int*)alloc((size_t)NEDGE * 4);
    int*    dstA = (int*)alloc((size_t)NEDGE * 4);
    float*  sc   = (float*)alloc((size_t)NEDGE * NHEAD * 4);
    float*  smax = (float*)alloc((size_t)NODES * NHEAD * 4);
    float*  ssum = (float*)alloc((size_t)NODES * NHEAD * 4);
    float*  hfin = (float*)alloc((size_t)NODES * HID * 4);
    float*  t1   = (float*)alloc((size_t)NODES * (HID / 2) * 4);

    auto blocks = [](int n) { return (n + 255) / 256; };
    const int mtiles      = NPAD / 32;                       // 313
    const int gemm_blocks = (mtiles * (HC / 32) + 7) / 8;    // 2504
    const dim3 tblk(32, 8);

    k_build_edges<<<blocks(NEDGE), 256, 0, stream>>>(ei, srcA, dstA);
    k_cvt_bf16<<<blocks(NODES * 512), 256, 0, stream>>>(x, XB, NODES * 512);

    for (int L = 0; L < 3; ++L) {
        const int Kin = (L == 0) ? 512 : HC;
        const dim3 tgrid(HC / 32, Kin / 32);
        // left/right transforms: hl = XB @ Wl + bl ; hr = XB @ Wr + br
        k_cvtT_bf16<<<tgrid, tblk, 0, stream>>>(Wl[L], WT, Kin);
        k_gemm_bf16<<<gemm_blocks, 256, 0, stream>>>(XB, WT, bl[L], hl, NODES, mtiles, Kin, HC);
        k_cvtT_bf16<<<tgrid, tblk, 0, stream>>>(Wr[L], WT2, Kin);
        k_gemm_bf16<<<gemm_blocks, 256, 0, stream>>>(XB, WT2, br[L], hr, NODES, mtiles, Kin, HC);

        // segment softmax + aggregation
        k_fill_f32<<<blocks(NODES * NHEAD), 256, 0, stream>>>(smax, -HUGE_VALF, NODES * NHEAD);
        k_fill_f32<<<blocks(NODES * NHEAD), 256, 0, stream>>>(ssum, 0.0f, NODES * NHEAD);
        k_fill_f32<<<blocks(NODES * HC), 256, 0, stream>>>(agg, 0.0f, NODES * HC);
        k_edge_score<<<blocks(NEDGE * NHEAD * 32), 256, 0, stream>>>(hl, hr, att[L], srcA, dstA, sc, smax);
        k_edge_exp<<<blocks(NEDGE * NHEAD), 256, 0, stream>>>(dstA, sc, smax, ssum);
        k_aggregate<<<blocks(NEDGE * HC), 256, 0, stream>>>(hl, srcA, dstA, sc, ssum, agg);

        if (L < 2)
            k_elu_bias_bf16<<<blocks(NODES * HC), 256, 0, stream>>>(agg, bb[L], XB);
        else
            k_mean_heads<<<blocks(NODES * HID), 256, 0, stream>>>(agg, bb[2], hfin);
    }

    // heads: pam50 = relu(h@Wp1+bp1)@Wp2+bp2 ; surv = relu(h@Ws1+bs1)@Ws2+bs2
    k_linear<<<blocks(NODES * 128), 256, 0, stream>>>(hfin, Wp1, bp1, t1, HID, 128, NODES * 128, 1);
    k_linear<<<blocks(NODES * 5),   256, 0, stream>>>(t1,   Wp2, bp2, outp, 128, 5, NODES * 5, 0);
    k_linear<<<blocks(NODES * 128), 256, 0, stream>>>(hfin, Ws1, bs1, t1, HID, 128, NODES * 128, 1);
    k_linear<<<blocks(NODES * 1),   256, 0, stream>>>(t1,   Ws2, bs2, outp + (size_t)NODES * 5, 128, 1, NODES, 0);
}
